// DGCNNSeg_76158360092795
// MI455X (gfx1250) — matrix operand order repaired and checked by reading, not compiled, not run on hardware
//
#include <hip/hip_runtime.h>

typedef __attribute__((ext_vector_type(16))) _Float16 v16h;
typedef __attribute__((ext_vector_type(8)))  _Float16 v8h;
typedef __attribute__((ext_vector_type(8)))  float    v8f;

constexpr int Bn = 2;
constexpr int Np = 8192;
constexpr int Kn = 32;

// ---------------------------------------------------------------------------
// helpers
// ---------------------------------------------------------------------------
__device__ __forceinline__ v8f wmma_f16(v16h a, v16h b, v8f c) {
  // D = A(16x32 f16) * B(32x16 f16) + C(16x16 f32)
  return __builtin_amdgcn_wmma_f32_16x16x32_f16(false, a, false, b, (short)0, c,
                                                false, false);
}

// async global -> LDS copy of one 16B unit (ASYNCcnt-tracked)
__device__ __forceinline__ void async_g2l_b128(const void* gsrc, void* ldst) {
  unsigned int lds = (unsigned int)(uintptr_t)ldst;   // low 32b = LDS byte addr
  asm volatile("global_load_async_to_lds_b128 %0, %1, off"
               :: "v"(lds), "v"(gsrc) : "memory");
}
__device__ __forceinline__ void async_wait0() {
  asm volatile("s_wait_asynccnt 0x0" ::: "memory");
}

// order-independent float atomic max via integer atomics (deterministic)
__device__ __forceinline__ void atomicMaxF(float* addr, float v) {
  if (v >= 0.0f) atomicMax((int*)addr, __float_as_int(v));
  else           atomicMin((unsigned int*)addr, __float_as_uint(v));
}

__device__ __forceinline__ float actf(float y, int act) {
  if (act == 1) return fmaxf(y, 0.0f);
  if (act == 2) return (y >= 0.0f) ? y : 0.01f * y;
  return y;
}

// ---------------------------------------------------------------------------
// tiny utility kernels
// ---------------------------------------------------------------------------
__global__ void k_fill(float* p, float v, int n) {
  int i = blockIdx.x * blockDim.x + threadIdx.x;
  if (i < n) p[i] = v;
}

__global__ void k_cvt(const float* s, _Float16* d, int n) {
  int i = blockIdx.x * blockDim.x + threadIdx.x;
  if (i < n) d[i] = (_Float16)s[i];
}

// pad w_ec1a (64,6) -> (64,32): cols 0..2 = nb-ctr weights, 16..18 = ctr weights
__global__ void k_pad_w1a(const float* w, _Float16* d) {
  int i = blockIdx.x * blockDim.x + threadIdx.x;
  if (i >= 64 * 32) return;
  int o = i >> 5, c = i & 31;
  float v = 0.0f;
  if (c < 3)                  v = w[o * 6 + c];
  else if (c >= 16 && c < 19) v = w[o * 6 + 3 + (c - 16)];
  d[i] = (_Float16)v;
}

// BN -> per-channel scale/shift: sc = g*rsqrt(v+eps); sh = b - m*sc
__global__ void k_bnprep(const float* g, const float* b, const float* m,
                         const float* v, float* sc, float* sh) {
  int i = blockIdx.x * blockDim.x + threadIdx.x;
  if (i >= 5 * 64) return;
  float s = g[i] * rsqrtf(v[i] + 1e-5f);
  sc[i] = s;
  sh[i] = b[i] - m[i] * s;
}

// ---------------------------------------------------------------------------
// generic WMMA GEMM:  D[m][o] = sum_k A[m][k] * W[o][k]
// A: (M,Kd) f16 row-major, W: (O,Kd) f16 row-major, Kd % 32 == 0,
// KC=min(Kd,128) must divide Kd (holds for 32/64/128/256/1280).
// Block = (32,8): 8 waves, each computes a 16x64 strip (4 WMMA / A-fetch).
// W strip (64 rows x KC) staged in LDS via GLOBAL_LOAD_ASYNC_TO_LDS_B128,
// shared by all 8 waves; LDS rows padded (+8 halves) against bank conflicts.
// epilogue: y = D*sc[o] + sh[o] (sc==null -> 1); act 0/1/2 = none/relu/lrelu
// mode 0: store OutH (f16) and/or OutF (f32); mode 2: atomicMax into
//         OutF[(row/rpg)*O + o]  (fused max-pool)
// launch: grid(M/128, O/64), block(32,8)
// ---------------------------------------------------------------------------
__global__ void k_gemm(const _Float16* __restrict__ A,
                       const _Float16* __restrict__ W,
                       const float* __restrict__ sc, const float* __restrict__ sh,
                       _Float16* OutH, float* OutF,
                       int Kd, int O, int act, int mode, int rpg) {
  __shared__ _Float16 smW[64 * 136];          // up to 64 x (128+8) halves
  int tid = threadIdx.y * 32 + threadIdx.x;
  int l = threadIdx.x;
  int lm = l & 15, hi = l >> 4;
  long m0 = ((long)blockIdx.x * 8 + threadIdx.y) * 16;
  int  o0 = blockIdx.y * 64;

  int KC = Kd < 128 ? Kd : 128;
  int SW = KC + 8;                            // padded LDS row stride (halves)

  const _Float16* arow = A + (size_t)(m0 + lm) * Kd;

  v8f cc[4] = {};
  for (int kc = 0; kc < Kd; kc += KC) {
    __syncthreads();                          // smW reusable
    for (int u = tid; u < 8 * KC; u += 256) { // 16B units: 64*KC*2/16 = 8*KC
      int row  = (u * 8) / KC;                // 8 halves per unit
      int colh = (u * 8) % KC;
      async_g2l_b128(W + (size_t)(o0 + row) * Kd + kc + colh,
                     &smW[row * SW + colh]);
    }
    async_wait0();
    __syncthreads();

    for (int k0 = kc; k0 < kc + KC; k0 += 32) {
      __builtin_prefetch(arow + k0 + 64, 0, 3);   // global_prefetch_b8 (near)
      // A lane layout: M = lane&15; halves 0..7 -> K=k0+hi*8; 8..15 -> +16
      v8h r0 = *(const v8h*)(arow + k0 + hi * 8);
      v8h r1 = *(const v8h*)(arow + k0 + 16 + hi * 8);
      v16h a;
#pragma unroll
      for (int i = 0; i < 8; i++) { a[i] = r0[i]; a[8 + i] = r1[i]; }
      int kk = k0 - kc;
#pragma unroll
      for (int t = 0; t < 4; t++) {
        // B lane layout: N = lane&15; halves 0..7 -> K=k0+hi*8; 8..15 -> +16
        const _Float16* ws = &smW[(t * 16 + lm) * SW + kk + hi * 8];
        v8h b0 = *(const v8h*)(ws);
        v8h b1 = *(const v8h*)(ws + 16);
        v16h bm;
#pragma unroll
        for (int i = 0; i < 8; i++) { bm[i] = b0[i]; bm[8 + i] = b1[i]; }
        cc[t] = wmma_f16(a, bm, cc[t]);
      }
    }
  }

#pragma unroll
  for (int t = 0; t < 4; t++) {
    int col = o0 + t * 16 + lm;
    float s = sc ? sc[col] : 1.0f;
    float sb = sh ? sh[col] : 0.0f;
#pragma unroll
    for (int i = 0; i < 8; i++) {
      long row = m0 + hi * 8 + i;
      float y = actf(cc[t][i] * s + sb, act);
      if (mode == 2) {
        atomicMaxF(OutF + (size_t)(row / rpg) * O + col, y);
      } else {
        if (OutH) OutH[(size_t)row * O + col] = (_Float16)y;
        if (OutF) OutF[(size_t)row * O + col] = y;
      }
    }
  }
}

// ---------------------------------------------------------------------------
// EdgeConv layer-a GEMM with fused edge-feature construction:
// rows r = edges (B,N,32); A[r][c] = c<Cf ? feat[nb][c]-feat[ctr][c]
//                                         : feat[ctr][c-Cf]
// feat: (B,N,Cf) f16 (Cf in {16,64}), idx: (B,N,32) int (per-batch indices)
// W: (O=64, 2*Cf) f16, fully staged in LDS (async). Wave = 16x64 strip.
// Epilogue = BN scale/shift + lrelu.
// mode 0: store OutH (edges,64) f16; mode 1: atomicMaxF into OutF[(r/32)*64+o]
// launch: grid(NE/128, 1), block(32,8)
// ---------------------------------------------------------------------------
__global__ void k_edge_gemm(const _Float16* __restrict__ feat,
                            const int* __restrict__ idx,
                            const _Float16* __restrict__ W,
                            const float* __restrict__ sc,
                            const float* __restrict__ sh,
                            _Float16* OutH, float* OutF,
                            int Cf, int act, int mode) {
  constexpr int O = 64;
  __shared__ _Float16 smW[64 * 136];
  int tid = threadIdx.y * 32 + threadIdx.x;
  int l = threadIdx.x;
  int lm = l & 15, hi = l >> 4;
  long m0 = ((long)blockIdx.x * 8 + threadIdx.y) * 16;

  int Kd = 2 * Cf;
  int SW = Kd + 8;

  // stage full W (64 x Kd) in LDS via async copy
  for (int u = tid; u < 8 * Kd; u += 256) {
    int row  = (u * 8) / Kd;
    int colh = (u * 8) % Kd;
    async_g2l_b128(W + (size_t)row * Kd + colh, &smW[row * SW + colh]);
  }

  long r = m0 + lm;                        // edge row this lane feeds A from
  int  b = (int)(r >> 18);                 // N*K = 262144 = 2^18
  int  p = (int)((r >> 5) & (Np - 1));
  int  nbi = idx[r];
  const _Float16* ctr = feat + ((size_t)b * Np + p)   * Cf;
  const _Float16* nb  = feat + ((size_t)b * Np + nbi) * Cf;

  async_wait0();
  __syncthreads();

  v8f cc[4] = {};
  for (int k0 = 0; k0 < Kd; k0 += 32) {
    int ka = k0 + hi * 8;
    int kb = k0 + 16 + hi * 8;
    v8h r0, r1;
    if (ka < Cf) { v8h nv = *(const v8h*)(nb + ka);
                   v8h cv = *(const v8h*)(ctr + ka); r0 = nv - cv; }
    else         { r0 = *(const v8h*)(ctr + ka - Cf); }
    if (kb < Cf) { v8h nv = *(const v8h*)(nb + kb);
                   v8h cv = *(const v8h*)(ctr + kb); r1 = nv - cv; }
    else         { r1 = *(const v8h*)(ctr + kb - Cf); }
    v16h a;
#pragma unroll
    for (int i = 0; i < 8; i++) { a[i] = r0[i]; a[8 + i] = r1[i]; }
#pragma unroll
    for (int t = 0; t < 4; t++) {
      const _Float16* ws = &smW[(t * 16 + lm) * SW + k0 + hi * 8];
      v8h b0 = *(const v8h*)(ws);
      v8h b1 = *(const v8h*)(ws + 16);
      v16h bm;
#pragma unroll
      for (int i = 0; i < 8; i++) { bm[i] = b0[i]; bm[8 + i] = b1[i]; }
      cc[t] = wmma_f16(a, bm, cc[t]);
    }
  }

#pragma unroll
  for (int t = 0; t < 4; t++) {
    int col = t * 16 + lm;
    float s = sc[col], sb = sh[col];
#pragma unroll
    for (int i = 0; i < 8; i++) {
      long row = m0 + hi * 8 + i;
      float y = actf(cc[t][i] * s + sb, act);
      if (mode == 0) OutH[(size_t)row * O + col] = (_Float16)y;
      else           atomicMaxF(OutF + (size_t)(row >> 5) * O + col, y);
    }
  }
}

// ---------------------------------------------------------------------------
// T-Net pointwise MLP 3 -> 64 -> 128 (relu), output f16 for the WMMA stage
// ---------------------------------------------------------------------------
__global__ void k_tnet_mlp(const float* __restrict__ x,
                           const float* __restrict__ tw1, const float* __restrict__ tb1,
                           const float* __restrict__ tw2, const float* __restrict__ tb2,
                           _Float16* __restrict__ h2f) {
  int i = blockIdx.x * blockDim.x + threadIdx.x;
  if (i >= Bn * Np) return;
  float x0 = x[i * 3 + 0], x1 = x[i * 3 + 1], x2 = x[i * 3 + 2];
  float h1[64];
#pragma unroll
  for (int o = 0; o < 64; o++) {
    float a = tw1[o * 3] * x0 + tw1[o * 3 + 1] * x1 + tw1[o * 3 + 2] * x2 + tb1[o];
    h1[o] = fmaxf(a, 0.0f);
  }
  for (int o = 0; o < 128; o++) {
    float a = tb2[o];
#pragma unroll
    for (int c = 0; c < 64; c++) a += h1[c] * tw2[o * 64 + c];
    h2f[(size_t)i * 128 + o] = (_Float16)fmaxf(a, 0.0f);
  }
}

// T-Net head: g(1024) -> 512 -> 256 -> 9, +I, write T to ws and to d_out tail
__global__ void k_tnet_fc(const float* __restrict__ g,
                          const float* tfw1, const float* tfb1,
                          const float* tfw2, const float* tfb2,
                          const float* tfw3, const float* tfb3,
                          float* Tbuf, float* Tout) {
  __shared__ float sg[1024];
  __shared__ float s5[512];
  __shared__ float s2[256];
  int b = blockIdx.x, t = threadIdx.x;
  sg[t] = g[b * 1024 + t];
  __syncthreads();
  if (t < 512) {
    float a = tfb1[t];
    for (int c = 0; c < 1024; c++) a += tfw1[t * 1024 + c] * sg[c];
    s5[t] = fmaxf(a, 0.0f);
  }
  __syncthreads();
  if (t < 256) {
    float a = tfb2[t];
    for (int c = 0; c < 512; c++) a += tfw2[t * 512 + c] * s5[c];
    s2[t] = fmaxf(a, 0.0f);
  }
  __syncthreads();
  if (t < 9) {
    float a = tfb3[t];
    for (int c = 0; c < 256; c++) a += tfw3[t * 256 + c] * s2[c];
    if (t == 0 || t == 4 || t == 8) a += 1.0f;
    Tbuf[b * 9 + t] = a;
    Tout[b * 9 + t] = a;
  }
}

// xt = x @ T ; store padded f16 features (B,N,16) + squared norms
__global__ void k_transform(const float* __restrict__ x, const float* __restrict__ Tbuf,
                            _Float16* __restrict__ feat3, float* __restrict__ sqn) {
  int i = blockIdx.x * blockDim.x + threadIdx.x;
  if (i >= Bn * Np) return;
  int b = i / Np;
  const float* T = Tbuf + b * 9;
  float x0 = x[i * 3 + 0], x1 = x[i * 3 + 1], x2 = x[i * 3 + 2];
  float y0 = x0 * T[0] + x1 * T[3] + x2 * T[6];
  float y1 = x0 * T[1] + x1 * T[4] + x2 * T[7];
  float y2 = x0 * T[2] + x1 * T[5] + x2 * T[8];
  _Float16* f = feat3 + (size_t)i * 16;
#pragma unroll
  for (int c = 0; c < 16; c++) f[c] = (_Float16)0.0f;
  f[0] = (_Float16)y0; f[1] = (_Float16)y1; f[2] = (_Float16)y2;
  sqn[i] = y0 * y0 + y1 * y1 + y2 * y2;
}

// feature post-pass: f32 -> f16 copy + squared norms for next KNN
__global__ void k_featpost(const float* __restrict__ xf, _Float16* __restrict__ xh,
                           float* __restrict__ sqn) {
  int i = blockIdx.x * blockDim.x + threadIdx.x;
  if (i >= Bn * Np) return;
  float s = 0.0f;
  for (int c = 0; c < 64; c++) {
    float v = xf[(size_t)i * 64 + c];
    xh[(size_t)i * 64 + c] = (_Float16)v;
    s += v * v;
  }
  sqn[i] = s;
}

// ---------------------------------------------------------------------------
// KNN top-32 per query; LDS candidate tiles; register top-k (unsorted + worst)
// score = |c|^2 - 2 q.c  (|q|^2 constant per row); includes self (as ref does)
// launch: grid(B*N/128), block(128)
// ---------------------------------------------------------------------------
template <int C>
__global__ void k_knn(const _Float16* __restrict__ feat,
                      const float* __restrict__ sqn, int* __restrict__ out) {
  constexpr int TILE = 128;
  __shared__ _Float16 tf[TILE * C];
  __shared__ float    tn[TILE];
  int q = blockIdx.x * blockDim.x + threadIdx.x;
  int b = q / Np;
  int base = b * Np;

  float qf[C];
#pragma unroll
  for (int c = 0; c < C; c++) qf[c] = (float)feat[(size_t)q * C + c];

  float td[32]; int ti[32];
#pragma unroll
  for (int s = 0; s < 32; s++) { td[s] = 1e30f; ti[s] = 0; }
  float wd = 1e30f;

  for (int jb = 0; jb < Np; jb += TILE) {
    __syncthreads();
    for (int j = threadIdx.x; j < TILE; j += blockDim.x) {
      tn[j] = sqn[base + jb + j];
      for (int c = 0; c < C; c++)
        tf[j * C + c] = feat[(size_t)(base + jb + j) * C + c];
    }
    __syncthreads();
    for (int j = 0; j < TILE; j++) {
      float d = tn[j];
#pragma unroll
      for (int c = 0; c < C; c++) d -= 2.0f * qf[c] * (float)tf[j * C + c];
      if (d < wd) {
        bool rep = false;
#pragma unroll
        for (int s = 0; s < 32; s++) {
          if (!rep && td[s] == wd) { td[s] = d; ti[s] = jb + j; rep = true; }
        }
        float nw = -1e30f;
#pragma unroll
        for (int s = 0; s < 32; s++) nw = fmaxf(nw, td[s]);
        wd = nw;
      }
    }
  }
#pragma unroll
  for (int s = 0; s < 32; s++) out[(size_t)q * Kn + s] = ti[s];
}

// ---------------------------------------------------------------------------
// embedding -> gfcat[b][1024..1087]
// ---------------------------------------------------------------------------
__global__ void k_emb(const int* __restrict__ jaw, const float* __restrict__ tab,
                      const float* __restrict__ wemb, const float* __restrict__ bemb,
                      float* __restrict__ gfcat) {
  int t = blockIdx.x * blockDim.x + threadIdx.x;
  if (t >= Bn * 64) return;
  int b = t / 64, o = t & 63;
  int j = jaw[b];
  float a = bemb[o];
  for (int c = 0; c < 64; c++) a += tab[j * 64 + c] * wemb[o * 64 + c];
  gfcat[b * 1088 + 1024 + o] = a;
}

// global max over points of [x1|x2|x3] -> catmax (B,192)
__global__ void k_gmax192(const float* x1, const float* x2, const float* x3,
                          float* catmax) {
  int t = blockIdx.x * blockDim.x + threadIdx.x;
  if (t >= Bn * 192) return;
  int b = t / 192, o = t % 192;
  const float* src = (o < 64) ? x1 : (o < 128) ? x2 : x3;
  int oo = o & 63;
  float m = -1e30f;
  for (int n = 0; n < Np; n++)
    m = fmaxf(m, src[((size_t)b * Np + n) * 64 + oo]);
  catmax[b * 192 + o] = m;
}

// gfcat[b][0..1023] = lrelu(catmax @ w1.T + b1)
__global__ void k_w1gf(const float* __restrict__ catmax, const float* __restrict__ w1,
                       const float* __restrict__ b1, float* __restrict__ gfcat) {
  __shared__ float scm[192];
  int b = blockIdx.x, t = threadIdx.x;
  if (t < 192) scm[t] = catmax[b * 192 + t];
  __syncthreads();
  float a = b1[t];
  for (int c = 0; c < 192; c++) a += w1[t * 192 + c] * scm[c];
  gfcat[b * 1088 + t] = (a >= 0.0f) ? a : 0.01f * a;
}

// full (B,N,1280) f16 = [gfcat broadcast (1088) | x1 | x2 | x3]
__global__ void k_concat_full(const float* __restrict__ gfcat,
                              const float* __restrict__ x1, const float* __restrict__ x2,
                              const float* __restrict__ x3, _Float16* __restrict__ full) {
  long i = (long)blockIdx.x * blockDim.x + threadIdx.x;
  if (i >= (long)Bn * Np * 1280) return;
  long r = i / 1280;
  int  c = (int)(i % 1280);
  int  b = (int)(r / Np);
  float v;
  if (c < 1088)       v = gfcat[b * 1088 + c];
  else if (c < 1152)  v = x1[r * 64 + (c - 1088)];
  else if (c < 1216)  v = x2[r * 64 + (c - 1152)];
  else                v = x3[r * 64 + (c - 1216)];
  full[i] = (_Float16)v;
}

// final 128 -> 17 (no activation), O=17 too small for WMMA
__global__ void k_w5out(const _Float16* __restrict__ h, const float* __restrict__ w5,
                        const float* __restrict__ b5, float* __restrict__ out) {
  long i = (long)blockIdx.x * blockDim.x + threadIdx.x;
  if (i >= (long)Bn * Np * 17) return;
  long r = i / 17;
  int  o = (int)(i % 17);
  float a = b5[o];
  for (int c = 0; c < 128; c++)
    a += (float)h[r * 128 + c] * w5[o * 128 + c];
  out[i] = a;
}

// ---------------------------------------------------------------------------
// launcher
// ---------------------------------------------------------------------------
extern "C" void kernel_launch(void* const* d_in, const int* in_sizes, int n_in,
                              void* d_out, int out_size, void* d_ws, size_t ws_size,
                              hipStream_t stream) {
  (void)in_sizes; (void)n_in; (void)out_size; (void)ws_size;

  const float* x     = (const float*)d_in[0];
  const int*   jaw   = (const int*)  d_in[1];
  const float* tw1   = (const float*)d_in[2];
  const float* tb1   = (const float*)d_in[3];
  const float* tw2   = (const float*)d_in[4];
  const float* tb2   = (const float*)d_in[5];
  const float* tw3   = (const float*)d_in[6];
  const float* tb3   = (const float*)d_in[7];
  const float* tfw1  = (const float*)d_in[8];
  const float* tfb1  = (const float*)d_in[9];
  const float* tfw2  = (const float*)d_in[10];
  const float* tfb2  = (const float*)d_in[11];
  const float* tfw3  = (const float*)d_in[12];
  const float* tfb3  = (const float*)d_in[13];
  const float* embt  = (const float*)d_in[14];
  const float* wemb  = (const float*)d_in[15];
  const float* bemb  = (const float*)d_in[16];
  const float* wec1a = (const float*)d_in[17];
  const float* wec1b = (const float*)d_in[18];
  const float* wec2a = (const float*)d_in[19];
  const float* wec2b = (const float*)d_in[20];
  const float* wec3a = (const float*)d_in[21];
  const float* bng   = (const float*)d_in[22];
  const float* bnb   = (const float*)d_in[23];
  const float* bnm   = (const float*)d_in[24];
  const float* bnv   = (const float*)d_in[25];
  const float* w1    = (const float*)d_in[26];
  const float* b1    = (const float*)d_in[27];
  const float* w2    = (const float*)d_in[28];
  const float* b2    = (const float*)d_in[29];
  const float* w3    = (const float*)d_in[30];
  const float* b3    = (const float*)d_in[31];
  const float* w4    = (const float*)d_in[32];
  const float* b4    = (const float*)d_in[33];
  const float* w5    = (const float*)d_in[34];
  const float* b5    = (const float*)d_in[35];

  float* dout = (float*)d_out;
  float* Tout = dout + (size_t)Bn * Np * 17;   // (out, T) concatenated

  // --- workspace carve-up ---
  size_t off = 0;
  auto alloc = [&](size_t bytes) -> void* {
    void* p = (char*)d_ws + off;
    off += (bytes + 255) & ~(size_t)255;
    return p;
  };
  _Float16* h2f    = (_Float16*)alloc((size_t)Bn * Np * 128 * 2);
  _Float16* feat3  = (_Float16*)alloc((size_t)Bn * Np * 16 * 2);
  float*    sqn    = (float*)   alloc((size_t)Bn * Np * 4);
  int*      idxb   = (int*)     alloc((size_t)Bn * Np * Kn * 4);
  float*    g1024  = (float*)   alloc((size_t)Bn * 1024 * 4);
  float*    Tbuf   = (float*)   alloc(256);
  float*    gfcat  = (float*)   alloc((size_t)Bn * 1088 * 4);
  float*    catmax = (float*)   alloc((size_t)Bn * 192 * 4);
  float*    x1f    = (float*)   alloc((size_t)Bn * Np * 64 * 4);
  float*    x2f    = (float*)   alloc((size_t)Bn * Np * 64 * 4);
  float*    x3f    = (float*)   alloc((size_t)Bn * Np * 64 * 4);
  _Float16* x1h    = (_Float16*)alloc((size_t)Bn * Np * 64 * 2);
  _Float16* x2h    = (_Float16*)alloc((size_t)Bn * Np * 64 * 2);
  _Float16* acth   = (_Float16*)alloc((size_t)Bn * Np * Kn * 64 * 2);
  _Float16* fullh  = (_Float16*)alloc((size_t)Bn * Np * 1280 * 2);
  _Float16* hah    = (_Float16*)alloc((size_t)Bn * Np * 256 * 2);
  _Float16* hbh    = (_Float16*)alloc((size_t)Bn * Np * 256 * 2);
  _Float16* hch    = (_Float16*)alloc((size_t)Bn * Np * 128 * 2);
  _Float16* tw3h   = (_Float16*)alloc(1024 * 128 * 2);
  _Float16* w1aph  = (_Float16*)alloc(64 * 32 * 2);
  _Float16* w1bh   = (_Float16*)alloc(64 * 64 * 2);
  _Float16* w2ah   = (_Float16*)alloc(64 * 128 * 2);
  _Float16* w2bh   = (_Float16*)alloc(64 * 64 * 2);
  _Float16* w3ah   = (_Float16*)alloc(64 * 128 * 2);
  _Float16* w2h    = (_Float16*)alloc((size_t)256 * 1280 * 2);
  _Float16* w3h    = (_Float16*)alloc(256 * 256 * 2);
  _Float16* w4h    = (_Float16*)alloc(128 * 256 * 2);
  float*    bnsc   = (float*)   alloc(5 * 64 * 4);
  float*    bnsh   = (float*)   alloc(5 * 64 * 4);

  const int NP2 = Bn * Np;           // 16384 points
  const int NE  = Bn * Np * Kn;      // 524288 edge rows
  const dim3 GB(32, 8);              // 8 waves per block, one 16x64 strip each

  // --- weight conversions / prep ---
  k_cvt<<<(1024 * 128 + 255) / 256, 256, 0, stream>>>(tw3, tw3h, 1024 * 128);
  k_cvt<<<(64 * 64 + 255) / 256, 256, 0, stream>>>(wec1b, w1bh, 64 * 64);
  k_cvt<<<(64 * 128 + 255) / 256, 256, 0, stream>>>(wec2a, w2ah, 64 * 128);
  k_cvt<<<(64 * 64 + 255) / 256, 256, 0, stream>>>(wec2b, w2bh, 64 * 64);
  k_cvt<<<(64 * 128 + 255) / 256, 256, 0, stream>>>(wec3a, w3ah, 64 * 128);
  k_cvt<<<(256 * 1280 + 255) / 256, 256, 0, stream>>>(w2, w2h, 256 * 1280);
  k_cvt<<<(256 * 256 + 255) / 256, 256, 0, stream>>>(w3, w3h, 256 * 256);
  k_cvt<<<(128 * 256 + 255) / 256, 256, 0, stream>>>(w4, w4h, 128 * 256);
  k_pad_w1a<<<(64 * 32 + 255) / 256, 256, 0, stream>>>(wec1a, w1aph);
  k_bnprep<<<(320 + 63) / 64, 64, 0, stream>>>(bng, bnb, bnm, bnv, bnsc, bnsh);

  // --- init atomic-max targets ---
  k_fill<<<(NP2 * 64 + 255) / 256, 256, 0, stream>>>(x1f, -1e30f, NP2 * 64);
  k_fill<<<(NP2 * 64 + 255) / 256, 256, 0, stream>>>(x2f, -1e30f, NP2 * 64);
  k_fill<<<(NP2 * 64 + 255) / 256, 256, 0, stream>>>(x3f, -1e30f, NP2 * 64);
  k_fill<<<(Bn * 1024 + 255) / 256, 256, 0, stream>>>(g1024, 0.0f, Bn * 1024);

  // --- T-Net ---
  k_tnet_mlp<<<NP2 / 256, 256, 0, stream>>>(x, tw1, tb1, tw2, tb2, h2f);
  // 128 -> 1024 WMMA GEMM with fused relu + max-pool over points
  k_gemm<<<dim3(NP2 / 128, 1024 / 64), GB, 0, stream>>>(
      h2f, tw3h, nullptr, tb3, nullptr, g1024, 128, 1024, 1, 2, Np);
  k_tnet_fc<<<Bn, 1024, 0, stream>>>(g1024, tfw1, tfb1, tfw2, tfb2, tfw3, tfb3,
                                     Tbuf, Tout);
  k_transform<<<NP2 / 256, 256, 0, stream>>>(x, Tbuf, feat3, sqn);

  // --- embedding (fills gfcat tail) ---
  k_emb<<<1, 128, 0, stream>>>(jaw, embt, wemb, bemb, gfcat);

  // --- EdgeConv 1 (Cf=16 padded from 3) ---
  k_knn<16><<<NP2 / 128, 128, 0, stream>>>(feat3, sqn, idxb);
  k_edge_gemm<<<dim3(NE / 128, 1), GB, 0, stream>>>(
      feat3, idxb, w1aph, bnsc + 0, bnsh + 0, acth, nullptr, 16, 2, 0);
  k_gemm<<<dim3(NE / 128, 1), GB, 0, stream>>>(
      acth, w1bh, bnsc + 64, bnsh + 64, nullptr, x1f, 64, 64, 2, 2, 32);
  k_featpost<<<NP2 / 128, 128, 0, stream>>>(x1f, x1h, sqn);

  // --- EdgeConv 2 ---
  k_knn<64><<<NP2 / 128, 128, 0, stream>>>(x1h, sqn, idxb);
  k_edge_gemm<<<dim3(NE / 128, 1), GB, 0, stream>>>(
      x1h, idxb, w2ah, bnsc + 128, bnsh + 128, acth, nullptr, 64, 2, 0);
  k_gemm<<<dim3(NE / 128, 1), GB, 0, stream>>>(
      acth, w2bh, bnsc + 192, bnsh + 192, nullptr, x2f, 64, 64, 2, 2, 32);
  k_featpost<<<NP2 / 128, 128, 0, stream>>>(x2f, x2h, sqn);

  // --- EdgeConv 3 (single layer, fused k-max) ---
  k_knn<64><<<NP2 / 128, 128, 0, stream>>>(x2h, sqn, idxb);
  k_edge_gemm<<<dim3(NE / 128, 1), GB, 0, stream>>>(
      x2h, idxb, w3ah, bnsc + 256, bnsh + 256, nullptr, x3f, 64, 2, 1);

  // --- global features + head ---
  k_gmax192<<<(Bn * 192 + 63) / 64, 64, 0, stream>>>(x1f, x2f, x3f, catmax);
  k_w1gf<<<Bn, 1024, 0, stream>>>(catmax, w1, b1, gfcat);
  k_concat_full<<<(int)(((long)NP2 * 1280 + 255) / 256), 256, 0, stream>>>(
      gfcat, x1f, x2f, x3f, fullh);

  k_gemm<<<dim3(NP2 / 128, 256 / 64), GB, 0, stream>>>(
      fullh, w2h, nullptr, b2, hah, nullptr, 1280, 256, 2, 0, 1);
  k_gemm<<<dim3(NP2 / 128, 256 / 64), GB, 0, stream>>>(
      hah, w3h, nullptr, b3, hbh, nullptr, 256, 256, 2, 0, 1);
  k_gemm<<<dim3(NP2 / 128, 128 / 64), GB, 0, stream>>>(
      hbh, w4h, nullptr, b4, hch, nullptr, 256, 128, 2, 0, 1);

  k_w5out<<<(int)(((long)NP2 * 17 + 127) / 128), 128, 0, stream>>>(hch, w5, b5, dout);
}
